// SelfAttention_584115552328
// MI455X (gfx1250) — matrix operand order repaired and checked by reading, not compiled, not run on hardware
//
#include <hip/hip_runtime.h>
#include <hip/hip_bf16.h>

// ---------------------------------------------------------------------------
// SAGAN self-attention fused for MI455X (gfx1250, wave32, WMMA 16x16x32 f16).
//   B=8, C=256, C8=32, N=64*64=4096
// Pipeline (all f16 storage, f32 accumulate):
//   1) xT[b][n][c]   = f16(x[b][c][n])               (transpose+convert)
//   2) Q[b][n][32]   = xT @ wg^T + bg  (WMMA)        (queries, A-layout friendly)
//      K[b][n][32]   = xT @ wf^T + bf  (WMMA)        (keys,   B-layout friendly)
//   3) V[b][c][n]    = wh @ x + bh     (WMMA)        (values, B-layout friendly)
//   4) flash attention per 16-query tile: S tile = 1 WMMA (K=32 full reduce),
//      online softmax via shfl_xor over 16-lane halves, P staged via LDS
//      (s_wait_dscnt) into A-layout, acc += P@V with 4 WMMAs/wave.
//      Inner loop is software-pipelined: next K tiles + current V tiles are
//      issued before the softmax VALU block so exp/shuffles hide load latency.
//      -> Op[b][n][c] f16
//   5) out = gamma*(wo @ Op^T + bo) + x (WMMA, fused residual)
// ---------------------------------------------------------------------------

typedef __attribute__((ext_vector_type(16))) _Float16 v16h;
typedef __attribute__((ext_vector_type(8)))  _Float16 v8h;
typedef __attribute__((ext_vector_type(8)))  float    v8f;

constexpr int BATCH = 8;
constexpr int CH    = 256;
constexpr int CQK   = 32;
constexpr int NN    = 4096;

union AFrag { v16h v; v8h h[2]; };

// A fragment: 16x32 f16, row-major source, row stride `ld` halfs.
// lane L: row m = L&15; 16 contiguous K=[16*hb, 16*hb+16), hb=L>>4.
__device__ inline AFrag load_a_frag(const _Float16* base, size_t ld, int lane) {
  int m = lane & 15, hb = lane >> 4;
  AFrag a;
  a.h[0] = *(const v8h*)(base + (size_t)m * ld + hb * 16);
  a.h[1] = *(const v8h*)(base + (size_t)m * ld + hb * 16 + 8);
  return a;
}

// B fragment: 32x16 f16 (KxN) stored column-major ([n][k], k contiguous, col
// stride `ld` halfs). lane L: col n = L&15, 16 contiguous k at 16*(L>>4).
__device__ inline v16h load_b_frag(const _Float16* base, size_t ld, int lane) {
  int n = lane & 15, hb = lane >> 4;
  return *(const v16h*)(base + (size_t)n * ld + hb * 16);
}

__device__ inline v8f wmma_f16(v16h a, v16h b, v8f c) {
  return __builtin_amdgcn_wmma_f32_16x16x32_f16(false, a, false, b, (short)0, c,
                                                false, false);
}

// reductions within a 16-lane half of the wave32 (xor masks 1..8 stay in-half)
__device__ inline float half_max(float v) {
  v = fmaxf(v, __shfl_xor(v, 1));
  v = fmaxf(v, __shfl_xor(v, 2));
  v = fmaxf(v, __shfl_xor(v, 4));
  v = fmaxf(v, __shfl_xor(v, 8));
  return v;
}
__device__ inline float half_sum(float v) {
  v += __shfl_xor(v, 1);
  v += __shfl_xor(v, 2);
  v += __shfl_xor(v, 4);
  v += __shfl_xor(v, 8);
  return v;
}

// ---------------------------------------------------------------------------
__global__ void cvt_f32_f16_kernel(const float* __restrict__ src,
                                   _Float16* __restrict__ dst, int n) {
  int i = blockIdx.x * blockDim.x + threadIdx.x;
  if (i < n) dst[i] = (_Float16)src[i];
}

__global__ void transpose_x_kernel(const float* __restrict__ x,
                                   _Float16* __restrict__ xT) {
  const size_t total = (size_t)BATCH * CH * NN;
  for (size_t idx = blockIdx.x * (size_t)blockDim.x + threadIdx.x; idx < total;
       idx += (size_t)gridDim.x * blockDim.x) {
    size_t b = idx / ((size_t)NN * CH);
    size_t r = idx % ((size_t)NN * CH);
    size_t n = r / CH, c = r % CH;
    xT[idx] = (_Float16)x[(b * CH + c) * NN + n];
  }
}

// ---------------------------------------------------------------------------
// Q/K projections: one wave per 16-row n-tile; 4 acc tiles (Q cols 0..31, K cols 0..31)
__global__ __launch_bounds__(128) void proj_qk_kernel(
    const _Float16* __restrict__ xT, const _Float16* __restrict__ wgh,
    const float* __restrict__ bg, const _Float16* __restrict__ wfh,
    const float* __restrict__ bf, _Float16* __restrict__ Qh,
    _Float16* __restrict__ Kh) {
  int lane = threadIdx.x & 31;
  int wave = blockIdx.x * 4 + (threadIdx.x >> 5);
  int b = wave >> 8, nt = wave & 255;  // 256 n-tiles per batch
  int m = lane & 15, hb = lane >> 4;
  const _Float16* xrow = xT + ((size_t)b * NN + nt * 16) * CH;
  v8f q0 = {}, q1 = {}, f0 = {}, f1 = {};
  for (int c = 0; c < CH; c += 32) {
    AFrag a = load_a_frag(xrow + c, CH, lane);
    v16h bg0 = load_b_frag(wgh + c, CH, lane);
    v16h bg1 = load_b_frag(wgh + (size_t)16 * CH + c, CH, lane);
    v16h bf0 = load_b_frag(wfh + c, CH, lane);
    v16h bf1 = load_b_frag(wfh + (size_t)16 * CH + c, CH, lane);
    q0 = wmma_f16(a.v, bg0, q0);
    q1 = wmma_f16(a.v, bg1, q1);
    f0 = wmma_f16(a.v, bf0, f0);
    f1 = wmma_f16(a.v, bf1, f1);
  }
  float bgl0 = bg[m], bgl1 = bg[16 + m];
  float bfl0 = bf[m], bfl1 = bf[16 + m];
  _Float16* qo = Qh + ((size_t)b * NN + nt * 16) * CQK;
  _Float16* ko = Kh + ((size_t)b * NN + nt * 16) * CQK;
#pragma unroll
  for (int i = 0; i < 8; i++) {
    size_t r = (size_t)(i + 8 * hb) * CQK;
    qo[r + m] = (_Float16)(q0[i] + bgl0);
    qo[r + 16 + m] = (_Float16)(q1[i] + bgl1);
    ko[r + m] = (_Float16)(f0[i] + bfl0);
    ko[r + 16 + m] = (_Float16)(f1[i] + bfl1);
  }
}

// ---------------------------------------------------------------------------
// V projection: wave = (b, o-tile of 16, n-group of 64); A=wh rows, B=xT cols
__global__ __launch_bounds__(128) void proj_v_kernel(
    const _Float16* __restrict__ xT, const _Float16* __restrict__ whh,
    const float* __restrict__ bh, _Float16* __restrict__ Vh) {
  int lane = threadIdx.x & 31;
  int wave = blockIdx.x * 4 + (threadIdx.x >> 5);
  int ng = wave & 63, ot = (wave >> 6) & 15, b = wave >> 10;
  int m = lane & 15, hb = lane >> 4;
  const _Float16* arow = whh + (size_t)(ot * 16) * CH;
  const _Float16* brow = xT + ((size_t)b * NN + ng * 64) * CH;
  v8f acc[4] = {};
  for (int c = 0; c < CH; c += 32) {
    AFrag a = load_a_frag(arow + c, CH, lane);
#pragma unroll
    for (int t = 0; t < 4; t++) {
      v16h bfr = load_b_frag(brow + (size_t)t * 16 * CH + c, CH, lane);
      acc[t] = wmma_f16(a.v, bfr, acc[t]);
    }
  }
  _Float16* vo = Vh + ((size_t)b * CH + ot * 16) * NN + ng * 64;
#pragma unroll
  for (int i = 0; i < 8; i++) {
    float bias = bh[ot * 16 + i + 8 * hb];
#pragma unroll
    for (int t = 0; t < 4; t++)
      vo[(size_t)(i + 8 * hb) * NN + t * 16 + m] = (_Float16)(acc[t][i] + bias);
  }
}

// ---------------------------------------------------------------------------
// Flash attention: block = (b, 16-query tile), 4 waves each owning 64 V cols.
// Software-pipelined: next K tiles and current V tiles are fetched before the
// softmax VALU block so exp/shuffles + the LDS P round-trip hide load latency.
__global__ __launch_bounds__(128) void attn_kernel(
    const _Float16* __restrict__ Qh, const _Float16* __restrict__ Kh,
    const _Float16* __restrict__ Vh, _Float16* __restrict__ Oph) {
  __shared__ __align__(16) _Float16 psh[4][16][32];  // per-wave P tile (D->A relayout)
  int lane = threadIdx.x & 31, w = threadIdx.x >> 5;
  int m = lane & 15, hb = lane >> 4;
  int b = blockIdx.x >> 8, qt = blockIdx.x & 255;
  const _Float16* qb = Qh + ((size_t)b * NN + qt * 16) * CQK;
  AFrag qa = load_a_frag(qb, CQK, lane);  // hoisted: Q fragment is loop-invariant
  const _Float16* kb = Kh + (size_t)b * NN * CQK;
  const _Float16* vb = Vh + (size_t)b * CH * (size_t)NN;
  int c0 = w * 64;
  v8f acc[4] = {};
  float mi[8], li[8];
#pragma unroll
  for (int i = 0; i < 8; i++) { mi[i] = -3.0e38f; li[i] = 0.f; }
  const v8f vzero = {};
  // prologue: first K tiles
  v16h k0f = load_b_frag(kb, CQK, lane);
  v16h k1f = load_b_frag(kb + (size_t)16 * CQK, CQK, lane);
  for (int j0 = 0; j0 < NN; j0 += 32) {
    v8f s0 = wmma_f16(qa.v, k0f, vzero);  // scores, full K=32 in one WMMA
    v8f s1 = wmma_f16(qa.v, k1f, vzero);
    // issue current V tiles now; consumed only after the softmax block below
    v16h vf[4];
#pragma unroll
    for (int t = 0; t < 4; t++)
      vf[t] = load_b_frag(vb + (size_t)(c0 + t * 16) * NN + j0, NN, lane);
    // issue next iteration's K tiles (wrap-indexed: always in-bounds, no branch)
    int jn = (j0 + 32) & (NN - 1);
    k0f = load_b_frag(kb + (size_t)jn * CQK, CQK, lane);
    k1f = load_b_frag(kb + (size_t)(jn + 16) * CQK, CQK, lane);
    // online softmax (row = i + 8*hb for this lane-half)
#pragma unroll
    for (int i = 0; i < 8; i++) {
      float cm = half_max(fmaxf(s0[i], s1[i]));
      float mn = fmaxf(mi[i], cm);
      float sc = __expf(mi[i] - mn);
      float p0 = __expf(s0[i] - mn);
      float p1 = __expf(s1[i] - mn);
      li[i] = li[i] * sc + half_sum(p0 + p1);
      mi[i] = mn;
      acc[0][i] *= sc; acc[1][i] *= sc; acc[2][i] *= sc; acc[3][i] *= sc;
      psh[w][i + 8 * hb][m] = (_Float16)p0;
      psh[w][i + 8 * hb][16 + m] = (_Float16)p1;
    }
    asm volatile("s_wait_dscnt 0x0" ::: "memory");  // P stores visible to A-frag loads
    AFrag pa = load_a_frag(&psh[w][0][0], 32, lane);
#pragma unroll
    for (int t = 0; t < 4; t++) acc[t] = wmma_f16(pa.v, vf[t], acc[t]);
    asm volatile("s_wait_dscnt 0x0" ::: "memory");  // P reads done before next overwrite
  }
  _Float16* op = Oph + ((size_t)b * NN + qt * 16) * CH + c0;
#pragma unroll
  for (int i = 0; i < 8; i++) {
    float inv = 1.f / li[i];
#pragma unroll
    for (int t = 0; t < 4; t++)
      op[(size_t)(i + 8 * hb) * CH + t * 16 + m] = (_Float16)(acc[t][i] * inv);
  }
}

// ---------------------------------------------------------------------------
// Output conv + bias + gamma + residual, fused.
__global__ __launch_bounds__(128) void final_kernel(
    const _Float16* __restrict__ Oph, const _Float16* __restrict__ woh,
    const float* __restrict__ bo, const float* __restrict__ x,
    const float* __restrict__ gamma, float* __restrict__ out) {
  int lane = threadIdx.x & 31;
  int wave = blockIdx.x * 4 + (threadIdx.x >> 5);
  int ng = wave & 63, ot = (wave >> 6) & 15, b = wave >> 10;
  int m = lane & 15, hb = lane >> 4;
  const _Float16* arow = woh + (size_t)(ot * 16) * CH;
  const _Float16* brow = Oph + ((size_t)b * NN + ng * 64) * CH;
  v8f acc[4] = {};
  for (int c = 0; c < CH; c += 32) {
    AFrag a = load_a_frag(arow + c, CH, lane);
#pragma unroll
    for (int t = 0; t < 4; t++) {
      v16h bfr = load_b_frag(brow + (size_t)t * 16 * CH + c, CH, lane);
      acc[t] = wmma_f16(a.v, bfr, acc[t]);
    }
  }
  float gm = gamma[0];
#pragma unroll
  for (int i = 0; i < 8; i++) {
    int o = ot * 16 + i + 8 * hb;
    float bias = bo[o];
#pragma unroll
    for (int t = 0; t < 4; t++) {
      size_t n = (size_t)ng * 64 + t * 16 + m;
      size_t idx = ((size_t)b * CH + o) * NN + n;
      out[idx] = gm * (acc[t][i] + bias) + x[idx];
    }
  }
}

// ---------------------------------------------------------------------------
extern "C" void kernel_launch(void* const* d_in, const int* in_sizes, int n_in,
                              void* d_out, int out_size, void* d_ws,
                              size_t ws_size, hipStream_t stream) {
  const float* x  = (const float*)d_in[0];
  const float* wf = (const float*)d_in[1];
  const float* bf = (const float*)d_in[2];
  const float* wg = (const float*)d_in[3];
  const float* bg = (const float*)d_in[4];
  const float* wh = (const float*)d_in[5];
  const float* bh = (const float*)d_in[6];
  const float* wo = (const float*)d_in[7];
  const float* bo = (const float*)d_in[8];
  const float* gamma = (const float*)d_in[9];
  float* out = (float*)d_out;

  char* ws = (char*)d_ws;
  size_t off = 0;
  auto alloc = [&](size_t bytes) -> char* {
    char* p = ws + off;
    off += (bytes + 255) & ~(size_t)255;
    return p;
  };
  _Float16* xT  = (_Float16*)alloc((size_t)BATCH * NN * CH * 2);   // 16 MB
  _Float16* Qh  = (_Float16*)alloc((size_t)BATCH * NN * CQK * 2);  //  2 MB
  _Float16* Kh  = (_Float16*)alloc((size_t)BATCH * NN * CQK * 2);  //  2 MB
  _Float16* Vh  = (_Float16*)alloc((size_t)BATCH * CH * NN * 2);   // 16 MB
  _Float16* Oph = (_Float16*)alloc((size_t)BATCH * NN * CH * 2);   // 16 MB
  _Float16* wgh = (_Float16*)alloc((size_t)CQK * CH * 2);
  _Float16* wfh = (_Float16*)alloc((size_t)CQK * CH * 2);
  _Float16* whh = (_Float16*)alloc((size_t)CH * CH * 2);
  _Float16* woh = (_Float16*)alloc((size_t)CH * CH * 2);

  // weight conversions
  cvt_f32_f16_kernel<<<(CQK * CH + 255) / 256, 256, 0, stream>>>(wg, wgh, CQK * CH);
  cvt_f32_f16_kernel<<<(CQK * CH + 255) / 256, 256, 0, stream>>>(wf, wfh, CQK * CH);
  cvt_f32_f16_kernel<<<(CH * CH + 255) / 256, 256, 0, stream>>>(wh, whh, CH * CH);
  cvt_f32_f16_kernel<<<(CH * CH + 255) / 256, 256, 0, stream>>>(wo, woh, CH * CH);
  // x transpose + f16 convert
  transpose_x_kernel<<<4096, 256, 0, stream>>>(x, xT);
  // projections
  proj_qk_kernel<<<512, 128, 0, stream>>>(xT, wgh, bg, wfh, bf, Qh, Kh);   // 2048 waves
  proj_v_kernel<<<2048, 128, 0, stream>>>(xT, whh, bh, Vh);                // 8192 waves
  // fused flash attention
  attn_kernel<<<BATCH * (NN / 16), 128, 0, stream>>>(Qh, Kh, Vh, Oph);     // 2048 blocks
  // output conv + residual
  final_kernel<<<2048, 128, 0, stream>>>(Oph, woh, bo, x, gamma, out);
}
